// FilterBankAttention_64879775973747
// MI455X (gfx1250) — compile-verified, matrix operand order checked by reading, not run on hardware
//
#include <hip/hip_runtime.h>
#include <cmath>

// ---------------------------------------------------------------------------
// FilterBankAttention for MI455X (gfx1250): f16 WMMA pipeline with ping-pong
// double-buffered k-loops (no register-rotation copies).
// pos_bias collapses to posb[h][|i-j|] (S-length LUT per head).
// ---------------------------------------------------------------------------

typedef _Float16 half_t;
typedef __attribute__((ext_vector_type(16))) _Float16 v16h;
typedef __attribute__((ext_vector_type(8)))  _Float16 v8h;
typedef __attribute__((ext_vector_type(8)))  float    v8f;

#define DIMX     1024
#define HEADS    16
#define HEAD_DIM 64
#define NFILT    4
#define NTAPS    101
#define RESN     512
#define BATCH    2
#define SEQ      2048
#define MTOT     (BATCH * SEQ)   // 4096 rows

// 16-bit A/B fragment for v_wmma_f32_16x16x32_f16, per ISA 7.12.2:
// lane&15 selects row (A) / column (B); lane>>4 selects K-half.
// VGPR0-3 hold K = g*8 + [0..7], VGPR4-7 hold K = 16 + g*8 + [0..7].
// Source must be contiguous in K -> two 16-byte loads.
static __device__ __forceinline__ v16h load_frag(const half_t* p, int g) {
  v8h lo = *(const v8h*)(p + g * 8);
  v8h hi = *(const v8h*)(p + 16 + g * 8);
  v16h f;
#pragma unroll
  for (int i = 0; i < 8; ++i) { f[i] = lo[i]; f[8 + i] = hi[i]; }
  return f;
}

#define WMMA_F16(a, b, c) \
  __builtin_amdgcn_wmma_f32_16x16x32_f16(false, (a), false, (b), (short)0, (c), false, false)

// ---------------------------------------------------------------------------
// f32 -> f16 elementwise convert
// ---------------------------------------------------------------------------
__global__ void fb_cvt_kernel(const float* __restrict__ in, half_t* __restrict__ out, int n) {
  int i = blockIdx.x * blockDim.x + threadIdx.x;
  if (i < n) out[i] = (half_t)in[i];
}

// ---------------------------------------------------------------------------
// Causal FIR of coeffs (101 taps) with proto ramp, time domain. table[r][f].
// ---------------------------------------------------------------------------
__global__ void fb_table_kernel(const float* __restrict__ coeffs, float* __restrict__ table) {
  int r = blockIdx.x * blockDim.x + threadIdx.x;
  if (r >= RESN) return;
  const float inv = 1.0f / (float)(RESN - 1);
  for (int f = 0; f < NFILT; ++f) {
    float acc = 0.0f;
    int tmax = (r < NTAPS - 1) ? r : (NTAPS - 1);
    for (int t = 0; t <= tmax; ++t) acc += coeffs[f * NTAPS + t] * ((float)(r - t) * inv);
    table[r * NFILT + f] = acc;
  }
}

// ---------------------------------------------------------------------------
// posb[h][d] = sum_f table[idx(d)][f] * Wpos[h][f], idx(d)=round(d/(S-1)*(R-1))
// ---------------------------------------------------------------------------
__global__ void fb_posb_kernel(const float* __restrict__ table, const float* __restrict__ Wpos,
                               float* __restrict__ posb) {
  int d = blockIdx.x * blockDim.x + threadIdx.x;
  if (d >= SEQ) return;
  float rel = (float)d / (float)(SEQ - 1);
  int idx = (int)roundf(rel * (float)(RESN - 1));
  idx = idx < 0 ? 0 : (idx > RESN - 1 ? RESN - 1 : idx);
  float t0 = table[idx * NFILT + 0], t1 = table[idx * NFILT + 1];
  float t2 = table[idx * NFILT + 2], t3 = table[idx * NFILT + 3];
  for (int h = 0; h < HEADS; ++h) {
    posb[h * SEQ + d] = t0 * Wpos[h * NFILT + 0] + t1 * Wpos[h * NFILT + 1] +
                        t2 * Wpos[h * NFILT + 2] + t3 * Wpos[h * NFILT + 3];
  }
}

// ---------------------------------------------------------------------------
// Shared 32x64-tile GEMM core: acc += A[32 x K] * B^T[64 x K] fragments,
// ping-pong double-buffered over K (no buffer-rotation register copies).
// A rows at arow0/arow1 (16 each), B columns are rows of wrow (contiguous K).
// ---------------------------------------------------------------------------
static __device__ __forceinline__ void gemm32x64_k(const half_t* arow0, const half_t* arow1,
                                                   const half_t* wrow, int K, int g,
                                                   v8f acc0[4], v8f acc1[4]) {
  v16h a0A = load_frag(arow0, g);
  v16h a1A = load_frag(arow1, g);
  v16h bA[4], bB[4];
#pragma unroll
  for (int t = 0; t < 4; ++t) bA[t] = load_frag(wrow + (size_t)t * 16 * DIMX, g);

  for (int k0 = 0; k0 < K; k0 += 64) {
    const int k1 = k0 + 32;                     // always < K (K multiple of 64)
    v16h a0B = load_frag(arow0 + k1, g);
    v16h a1B = load_frag(arow1 + k1, g);
#pragma unroll
    for (int t = 0; t < 4; ++t) bB[t] = load_frag(wrow + (size_t)t * 16 * DIMX + k1, g);
#pragma unroll
    for (int t = 0; t < 4; ++t) {
      acc0[t] = WMMA_F16(a0A, bA[t], acc0[t]);
      acc1[t] = WMMA_F16(a1A, bA[t], acc1[t]);
    }
    const int k2 = (k0 + 64 < K) ? (k0 + 64) : k0;  // clamp final prefetch
    a0A = load_frag(arow0 + k2, g);
    a1A = load_frag(arow1 + k2, g);
#pragma unroll
    for (int t = 0; t < 4; ++t) bA[t] = load_frag(wrow + (size_t)t * 16 * DIMX + k2, g);
#pragma unroll
    for (int t = 0; t < 4; ++t) {
      acc0[t] = WMMA_F16(a0B, bB[t], acc0[t]);
      acc1[t] = WMMA_F16(a1B, bB[t], acc1[t]);
    }
  }
}

// ---------------------------------------------------------------------------
// QKV GEMM: qkv[s,e] = sum_d x[s,d] * Wqkv[e,d]. 32x64 tile per wave.
// Epilogue scatters: q*scale -> [b,h,s,dh], k -> [b,h,j,dh], v -> [b,h,dh,j].
// ---------------------------------------------------------------------------
__global__ __launch_bounds__(32) void fb_qkv_kernel(const half_t* __restrict__ Xh,
                                                    const half_t* __restrict__ Wh,
                                                    half_t* __restrict__ qh,
                                                    half_t* __restrict__ kh,
                                                    half_t* __restrict__ vt) {
  const int lane = threadIdx.x;
  const int g = lane >> 4, n = lane & 15;
  const int row0 = blockIdx.y * 32;
  const int e0 = blockIdx.x * 64;
  v8f acc0[4] = {v8f{}, v8f{}, v8f{}, v8f{}};
  v8f acc1[4] = {v8f{}, v8f{}, v8f{}, v8f{}};
  const half_t* arow0 = Xh + (size_t)(row0 + n) * DIMX;
  const half_t* arow1 = arow0 + (size_t)16 * DIMX;
  const half_t* wrow  = Wh + (size_t)(e0 + n) * DIMX;

  gemm32x64_k(arow0, arow1, wrow, DIMX, g, acc0, acc1);

  const float scale = 0.125f;  // HEAD_DIM^-0.5
#pragma unroll
  for (int p = 0; p < 2; ++p) {
#pragma unroll
    for (int t = 0; t < 4; ++t) {
      int e = e0 + t * 16 + n;
#pragma unroll
      for (int r = 0; r < 8; ++r) {
        int s = row0 + p * 16 + r + 8 * g;
        int bb = s >> 11, sl = s & (SEQ - 1);
        float val = p ? acc1[t][r] : acc0[t][r];
        if (e < DIMX) {
          int h = e >> 6, dh = e & 63;
          qh[(((size_t)(bb * HEADS + h)) * SEQ + sl) * HEAD_DIM + dh] = (half_t)(val * scale);
        } else if (e < 2 * DIMX) {
          int e2 = e - DIMX; int h = e2 >> 6, dh = e2 & 63;
          kh[(((size_t)(bb * HEADS + h)) * SEQ + sl) * HEAD_DIM + dh] = (half_t)val;
        } else {
          int e2 = e - 2 * DIMX; int h = e2 >> 6, dh = e2 & 63;
          vt[(((size_t)(bb * HEADS + h)) * HEAD_DIM + dh) * SEQ + sl] = (half_t)val;
        }
      }
    }
  }
}

// ---------------------------------------------------------------------------
// Attention: one wave per (b*h, 16-row strip).
// Pass1: logits (QK^T via WMMA, ping-pong over key tiles) + posb(LDS LUT).
// Pass2: row softmax with 128-bit LDS scans, unnormalized exp in place.
// Pass3: AV via WMMA (A from LDS, B from vt, ping-pong), 1/rowsum in epilogue.
// ---------------------------------------------------------------------------
#define SSTR 2056  // 2048 + 8 halves: bank-conflict padding, keeps 16B align

__global__ __launch_bounds__(32) void fb_attn_kernel(const half_t* __restrict__ qh,
                                                     const half_t* __restrict__ kh,
                                                     const half_t* __restrict__ vt,
                                                     const float* __restrict__ posb,
                                                     half_t* __restrict__ aoh) {
  __shared__ __align__(16) half_t lat[16 * SSTR];
  __shared__ __align__(16) float pbs[SEQ];
  __shared__ float rstat[16];
  const int lane = threadIdx.x;
  const int g = lane >> 4, n = lane & 15;
  const int bh = blockIdx.y;
  const int hh = bh & (HEADS - 1);
  const int i0 = blockIdx.x * 16;
  const half_t* qbase = qh + ((size_t)bh * SEQ + i0) * HEAD_DIM;
  const half_t* kbase = kh + (size_t)bh * SEQ * HEAD_DIM;
  const half_t* vbase = vt + (size_t)bh * HEAD_DIM * SEQ;
  const float* pb = posb + hh * SEQ;

  // Stage this head's posb row into LDS (float4 loads, conflict-free).
  for (int t = lane * 4; t < SEQ; t += 32 * 4)
    *(float4*)&pbs[t] = *(const float4*)&pb[t];

  // Q strip fragments (K = 0..31 and 32..63), reused across all key tiles.
  v16h aq0 = load_frag(qbase + n * HEAD_DIM + 0,  g);
  v16h aq1 = load_frag(qbase + n * HEAD_DIM + 32, g);
  __syncthreads();

  // ---- Pass 1: logits, ping-pong over key tiles (j-step 16, unrolled x2) ---
  v16h b0A = load_frag(kbase + (size_t)n * HEAD_DIM + 0,  g);
  v16h b1A = load_frag(kbase + (size_t)n * HEAD_DIM + 32, g);
  for (int j0 = 0; j0 < SEQ; j0 += 32) {
    const int j1 = j0 + 16;                     // always < SEQ
    v16h b0B = load_frag(kbase + (size_t)(j1 + n) * HEAD_DIM + 0,  g);
    v16h b1B = load_frag(kbase + (size_t)(j1 + n) * HEAD_DIM + 32, g);
    {
      v8f c = {};
      c = WMMA_F16(aq0, b0A, c);
      c = WMMA_F16(aq1, b1A, c);
      int j = j0 + n;
#pragma unroll
      for (int r = 0; r < 8; ++r) {
        int i = i0 + r + 8 * g;
        int di = i - j; if (di < 0) di = -di;
        lat[(r + 8 * g) * SSTR + j] = (half_t)(c[r] + pbs[di]);
      }
    }
    const int j2 = (j0 + 32 < SEQ) ? (j0 + 32) : j0;  // clamp final prefetch
    b0A = load_frag(kbase + (size_t)(j2 + n) * HEAD_DIM + 0,  g);
    b1A = load_frag(kbase + (size_t)(j2 + n) * HEAD_DIM + 32, g);
    {
      v8f c = {};
      c = WMMA_F16(aq0, b0B, c);
      c = WMMA_F16(aq1, b1B, c);
      int j = j1 + n;
#pragma unroll
      for (int r = 0; r < 8; ++r) {
        int i = i0 + r + 8 * g;
        int di = i - j; if (di < 0) di = -di;
        lat[(r + 8 * g) * SSTR + j] = (half_t)(c[r] + pbs[di]);
      }
    }
  }
  __syncthreads();

  // ---- Pass 2: softmax; lane handles row n, column half g (2 lanes/row) ----
  const int row = n;
  const int c0 = g * (SEQ / 2);
  float mx = -3.0e38f;
  for (int j = 0; j < SEQ / 2; j += 8) {
    v8h v = *(const v8h*)&lat[row * SSTR + c0 + j];
#pragma unroll
    for (int u = 0; u < 8; ++u) mx = fmaxf(mx, (float)v[u]);
  }
  mx = fmaxf(mx, __shfl_xor(mx, 16));
  float sum = 0.0f;
  for (int j = 0; j < SEQ / 2; j += 8) {
    v8h v = *(const v8h*)&lat[row * SSTR + c0 + j];
    v8h e;
#pragma unroll
    for (int u = 0; u < 8; ++u) {
      float ev = __expf((float)v[u] - mx);
      sum += ev;
      e[u] = (half_t)ev;
    }
    *(v8h*)&lat[row * SSTR + c0 + j] = e;  // in place: same lane owns same cells
  }
  sum += __shfl_xor(sum, 16);
  if (lane < 16) rstat[lane] = 1.0f / sum;
  __syncthreads();

  // ---- Pass 3: AV, ping-pong over K (k-step 32, unrolled x2) ---------------
  v8f o[4] = {v8f{}, v8f{}, v8f{}, v8f{}};
  v16h aA = load_frag(&lat[row * SSTR], g);
  v16h bvA[4], bvB[4];
#pragma unroll
  for (int t = 0; t < 4; ++t) bvA[t] = load_frag(vbase + (size_t)(t * 16 + n) * SEQ, g);
  for (int k0 = 0; k0 < SEQ; k0 += 64) {
    const int k1 = k0 + 32;                     // always < SEQ
    v16h aB = load_frag(&lat[row * SSTR + k1], g);
#pragma unroll
    for (int t = 0; t < 4; ++t) bvB[t] = load_frag(vbase + (size_t)(t * 16 + n) * SEQ + k1, g);
#pragma unroll
    for (int t = 0; t < 4; ++t) o[t] = WMMA_F16(aA, bvA[t], o[t]);
    const int k2 = (k0 + 64 < SEQ) ? (k0 + 64) : k0;  // clamp final prefetch
    aA = load_frag(&lat[row * SSTR + k2], g);
#pragma unroll
    for (int t = 0; t < 4; ++t) bvA[t] = load_frag(vbase + (size_t)(t * 16 + n) * SEQ + k2, g);
#pragma unroll
    for (int t = 0; t < 4; ++t) o[t] = WMMA_F16(aB, bvB[t], o[t]);
  }
  const int bidx = bh >> 4;
#pragma unroll
  for (int t = 0; t < 4; ++t) {
#pragma unroll
    for (int r = 0; r < 8; ++r) {
      int i = i0 + r + 8 * g;
      float val = o[t][r] * rstat[r + 8 * g];
      int dh = t * 16 + n;
      aoh[((size_t)(bidx * SEQ + i)) * DIMX + hh * HEAD_DIM + dh] = (half_t)val;
    }
  }
}

// ---------------------------------------------------------------------------
// Output projection: out[s,e] = sum_d ao[s,d] * Wout[e,d], f32 result.
// 32x64 tile per wave, same ping-pong GEMM core.
// ---------------------------------------------------------------------------
__global__ __launch_bounds__(32) void fb_out_kernel(const half_t* __restrict__ Ah,
                                                    const half_t* __restrict__ Wh,
                                                    float* __restrict__ out) {
  const int lane = threadIdx.x;
  const int g = lane >> 4, n = lane & 15;
  const int row0 = blockIdx.y * 32;
  const int e0 = blockIdx.x * 64;
  v8f acc0[4] = {v8f{}, v8f{}, v8f{}, v8f{}};
  v8f acc1[4] = {v8f{}, v8f{}, v8f{}, v8f{}};
  const half_t* arow0 = Ah + (size_t)(row0 + n) * DIMX;
  const half_t* arow1 = arow0 + (size_t)16 * DIMX;
  const half_t* wrow  = Wh + (size_t)(e0 + n) * DIMX;

  gemm32x64_k(arow0, arow1, wrow, DIMX, g, acc0, acc1);

#pragma unroll
  for (int p = 0; p < 2; ++p)
#pragma unroll
    for (int t = 0; t < 4; ++t)
#pragma unroll
      for (int r = 0; r < 8; ++r) {
        int s = row0 + p * 16 + r + 8 * g;
        out[(size_t)s * DIMX + e0 + t * 16 + n] = p ? acc1[t][r] : acc0[t][r];
      }
}

// ---------------------------------------------------------------------------
extern "C" void kernel_launch(void* const* d_in, const int* in_sizes, int n_in,
                              void* d_out, int out_size, void* d_ws, size_t ws_size,
                              hipStream_t stream) {
  const float* x      = (const float*)d_in[0];  // (2,2048,1024)
  const float* Wqkv   = (const float*)d_in[1];  // (3072,1024)
  const float* Wout   = (const float*)d_in[2];  // (1024,1024)
  const float* Wpos   = (const float*)d_in[3];  // (16,4)
  const float* coeffs = (const float*)d_in[4];  // (4,101)
  (void)in_sizes; (void)n_in; (void)out_size; (void)ws_size;

  char* ws = (char*)d_ws;
  size_t off = 0;
  auto take = [&](size_t bytes) -> char* {
    char* p = ws + off;
    off = (off + bytes + 255) & ~(size_t)255;
    return p;
  };
  float*  table = (float*) take((size_t)RESN * NFILT * sizeof(float));
  float*  posb  = (float*) take((size_t)HEADS * SEQ * sizeof(float));
  half_t* xh    = (half_t*)take((size_t)MTOT * DIMX * sizeof(half_t));
  half_t* wqh   = (half_t*)take((size_t)3 * DIMX * DIMX * sizeof(half_t));
  half_t* woh   = (half_t*)take((size_t)DIMX * DIMX * sizeof(half_t));
  half_t* qh    = (half_t*)take((size_t)MTOT * DIMX * sizeof(half_t));
  half_t* kh    = (half_t*)take((size_t)MTOT * DIMX * sizeof(half_t));
  half_t* vth   = (half_t*)take((size_t)MTOT * DIMX * sizeof(half_t));
  half_t* aoh   = (half_t*)take((size_t)MTOT * DIMX * sizeof(half_t));

  const int nx = MTOT * DIMX;
  fb_cvt_kernel<<<(nx + 255) / 256, 256, 0, stream>>>(x, xh, nx);
  const int nq = 3 * DIMX * DIMX;
  fb_cvt_kernel<<<(nq + 255) / 256, 256, 0, stream>>>(Wqkv, wqh, nq);
  const int nw = DIMX * DIMX;
  fb_cvt_kernel<<<(nw + 255) / 256, 256, 0, stream>>>(Wout, woh, nw);

  fb_table_kernel<<<(RESN + 255) / 256, 256, 0, stream>>>(coeffs, table);
  fb_posb_kernel<<<(SEQ + 255) / 256, 256, 0, stream>>>(table, Wpos, posb);

  fb_qkv_kernel<<<dim3(3 * DIMX / 64, MTOT / 32), 32, 0, stream>>>(xh, wqh, qh, kh, vth);
  fb_attn_kernel<<<dim3(SEQ / 16, BATCH * HEADS), 32, 0, stream>>>(qh, kh, vth, posb, aoh);
  fb_out_kernel<<<dim3(DIMX / 64, MTOT / 32), 32, 0, stream>>>(aoh, woh, (float*)d_out);
}
